// InferenceLTMPBlock_42030549959154
// MI455X (gfx1250) — compile-verified, matrix operand order checked by reading, not run on hardware
//
#include <hip/hip_runtime.h>
#include <hip/hip_bf16.h>
#include <math.h>

// ---------------------------------------------------------------------------
// CDNA5 (gfx1250) wave32 WMMA bf16 building blocks
// ---------------------------------------------------------------------------
typedef __attribute__((ext_vector_type(16))) __bf16 v16bf;
typedef __attribute__((ext_vector_type(8)))  float  v8f;

union BFrag {
  v16bf v;
  uint4 q[2];
  unsigned short s[16];
};

__device__ __forceinline__ unsigned short f2bf(float f) {
  unsigned int u = __float_as_uint(f);
  u += 0x7fffu + ((u >> 16) & 1u);           // round-to-nearest-even
  return (unsigned short)(u >> 16);
}
__device__ __forceinline__ float bf2f(unsigned short s) {
  return __uint_as_float(((unsigned int)s) << 16);
}

// 16-bit A/B fragment, row-major source. Per ISA 7.12.2: lane l holds row
// (l&15); elements 0..7 = K[khalf..khalf+7], elements 8..15 = K[khalf+16..+23],
// khalf = 8*(l>=16). Two contiguous 16B chunks -> global_load_b128.
__device__ __forceinline__ v16bf load_frag(const unsigned short* rowPtr, int khalf) {
  BFrag f;
  f.q[0] = *(const uint4*)(rowPtr + khalf);
  f.q[1] = *(const uint4*)(rowPtr + khalf + 16);
  return f.v;
}
__device__ __forceinline__ v8f wmma_bf16(v16bf a, v16bf b, v8f c) {
  return __builtin_amdgcn_wmma_f32_16x16x32_bf16(false, a, false, b, (short)0, c,
                                                 false, false);
}

// ---------------------------------------------------------------------------
// Elementwise / LN kernels
// ---------------------------------------------------------------------------
__global__ void k_f2b(const float* __restrict__ in, unsigned short* __restrict__ out, int n) {
  int i = blockIdx.x * blockDim.x + threadIdx.x;
  if (i < n) out[i] = f2bf(in[i]);
}

__global__ void k_zero(float* p, int n) {
  int i = blockIdx.x * blockDim.x + threadIdx.x;
  if (i < n) p[i] = 0.f;
}

__global__ __launch_bounds__(256)
void k_layernorm(const float* __restrict__ x, const float* __restrict__ g,
                 const float* __restrict__ b, unsigned short* __restrict__ out, int C) {
  int row = blockIdx.x, tid = threadIdx.x;
  const float* xr = x + (size_t)row * C;
  __shared__ float red[256];
  float s = 0.f;
  for (int c = tid; c < C; c += 256) s += xr[c];
  red[tid] = s; __syncthreads();
  for (int o = 128; o > 0; o >>= 1) { if (tid < o) red[tid] += red[tid + o]; __syncthreads(); }
  float mean = red[0] / C;
  __syncthreads();
  float v = 0.f;
  for (int c = tid; c < C; c += 256) { float d = xr[c] - mean; v += d * d; }
  red[tid] = v; __syncthreads();
  for (int o = 128; o > 0; o >>= 1) { if (tid < o) red[tid] += red[tid + o]; __syncthreads(); }
  float rstd = rsqrtf(red[0] / C + 1e-5f);
  for (int c = tid; c < C; c += 256)
    out[(size_t)row * C + c] = f2bf((xr[c] - mean) * rstd * g[c] + b[c]);
}

// ---------------------------------------------------------------------------
// Generic bf16 WMMA GEMM: out[M,N] = A[M,K] @ W[N,K]^T (+bias)(+gelu)(+resid)
// Block = 256 threads = 8 waves; tile 128(M) x 64(N); wave tile 16x64.
// A fragment reused across the wave's 4 N-subtiles; W reuse comes from the
// 192MB L2 (all weights here fit), so no LDS staging needed.
// ---------------------------------------------------------------------------
__global__ __launch_bounds__(256)
void k_gemm(const unsigned short* __restrict__ A, const unsigned short* __restrict__ W,
            int M, int N, int K,
            const float* __restrict__ bias, const float* __restrict__ resid,
            int gelu, float* __restrict__ outF, unsigned short* __restrict__ outB) {
  int lane  = threadIdx.x & 31;
  int wv    = threadIdx.x >> 5;
  int m0    = blockIdx.y * 128 + wv * 16;
  int n0    = blockIdx.x * 64;
  int r     = lane & 15;
  int khalf = (lane >> 4) * 8;

  const unsigned short* arow  = A + (size_t)(m0 + r) * K;
  const unsigned short* wrow0 = W + (size_t)(n0 +  0 + r) * K;
  const unsigned short* wrow1 = W + (size_t)(n0 + 16 + r) * K;
  const unsigned short* wrow2 = W + (size_t)(n0 + 32 + r) * K;
  const unsigned short* wrow3 = W + (size_t)(n0 + 48 + r) * K;

  v8f acc0 = {}, acc1 = {}, acc2 = {}, acc3 = {};
  for (int k0 = 0; k0 < K; k0 += 32) {
    v16bf a = load_frag(arow + k0, khalf);
    __builtin_prefetch(arow + k0 + 128, 0, 1);    // global_prefetch_b8
    __builtin_prefetch(wrow0 + k0 + 128, 0, 1);
    acc0 = wmma_bf16(a, load_frag(wrow0 + k0, khalf), acc0);
    acc1 = wmma_bf16(a, load_frag(wrow1 + k0, khalf), acc1);
    acc2 = wmma_bf16(a, load_frag(wrow2 + k0, khalf), acc2);
    acc3 = wmma_bf16(a, load_frag(wrow3 + k0, khalf), acc3);
  }
  v8f accv[4] = {acc0, acc1, acc2, acc3};
#pragma unroll
  for (int t = 0; t < 4; t++) {
    int col = n0 + t * 16 + r;
    float bv = bias ? bias[col] : 0.f;
#pragma unroll
    for (int v = 0; v < 8; v++) {
      int row = m0 + v + 8 * (lane >> 4);
      float xv = accv[t][v] + bv;
      if (gelu) xv = 0.5f * xv * (1.0f + erff(xv * 0.70710678118654752f));
      if (resid) xv += resid[(size_t)row * N + col];
      if (outF) outF[(size_t)row * N + col] = xv;
      if (outB) outB[(size_t)row * N + col] = f2bf(xv);
    }
  }
}

// ---------------------------------------------------------------------------
// V transpose: vt[(b*12+h)*64 + d][n] = V[b,n,h,d].  One-time 12.6MB buffer;
// lives in L2, turns every P@V B-fragment into two b128 loads per lane.
// ---------------------------------------------------------------------------
__global__ void k_vtrans(const unsigned short* __restrict__ qkv,
                         unsigned short* __restrict__ vt) {
  int i = blockIdx.x * blockDim.x + threadIdx.x;   // (((b*12+h)*64)+d)*1024 + n
  if (i >= 8 * 12 * 64 * 1024) return;
  int n = i & 1023;
  int d = (i >> 10) & 63;
  int h = (i >> 16) % 12;
  int b = i / (12 * 64 * 1024);
  vt[i] = qkv[(size_t)(b * 1024 + n) * 2304 + 1536 + h * 64 + d];
}

// ---------------------------------------------------------------------------
// Attention pass A: per-row logsumexp of (Q.K^T * scale + log(size[k])).
// One wave per (b, h, 16-query tile). wave32 C-layout reduced via shfl_xor.
// ---------------------------------------------------------------------------
__global__ __launch_bounds__(256)
void k_attn_lse(const unsigned short* __restrict__ qkv, const float* __restrict__ size,
                float* __restrict__ lse) {
  const int N = 1024, H = 12, TC = 2304;
  const float scale = 0.125f;  // 64^-0.5
  int gw   = (blockIdx.x * blockDim.x + threadIdx.x) >> 5;
  int lane = threadIdx.x & 31;
  int qt   = gw & 63;
  int h    = (gw >> 6) % H;
  int b    = gw / (64 * H);
  int r = lane & 15, khalf = (lane >> 4) * 8;

  const unsigned short* qrow = qkv + (size_t)(b * N + qt * 16 + r) * TC + h * 64;
  v16bf qa0 = load_frag(qrow, khalf);
  v16bf qa1 = load_frag(qrow + 32, khalf);

  float rm[8], rs[8];
#pragma unroll
  for (int v = 0; v < 8; v++) { rm[v] = -INFINITY; rs[v] = 0.f; }

  for (int kt = 0; kt < 64; kt++) {
    const unsigned short* krow = qkv + (size_t)(b * N + kt * 16 + r) * TC + 768 + h * 64;
    v8f c = {};
    c = wmma_bf16(qa0, load_frag(krow, khalf), c);
    c = wmma_bf16(qa1, load_frag(krow + 32, khalf), c);
    float lsz = __logf(size[b * N + kt * 16 + r]);
#pragma unroll
    for (int v = 0; v < 8; v++) {
      float s  = c[v] * scale + lsz;
      float nm = fmaxf(rm[v], s);
      rs[v] = rs[v] * __expf(rm[v] - nm) + __expf(s - nm);
      rm[v] = nm;
    }
  }
#pragma unroll
  for (int v = 0; v < 8; v++) {
    float m = rm[v], s = rs[v];
    for (int off = 1; off < 16; off <<= 1) {          // reduce within lane groups 0-15 / 16-31
      float m2 = __shfl_xor(m, off, 32);
      float s2 = __shfl_xor(s, off, 32);
      float M  = fmaxf(m, m2);
      s = s * __expf(m - M) + s2 * __expf(m2 - M);
      m = M;
    }
    if (r == 0)
      lse[((size_t)(b * H + h)) * N + qt * 16 + v + 8 * (lane >> 4)] = m + __logf(s);
  }
}

// ---------------------------------------------------------------------------
// Attention pass B: P = exp(S - lse); xo = P @ V (WMMA); importance += sum_q P.
// P re-laid out C->A through per-wave LDS tile (explicit s_wait_dscnt).
// V fragments come from the pre-transposed vt buffer -> b128 loads.
// ---------------------------------------------------------------------------
__global__ __launch_bounds__(256)
void k_attn_av(const unsigned short* __restrict__ qkv, const unsigned short* __restrict__ vt,
               const float* __restrict__ size, const float* __restrict__ lse,
               float* __restrict__ imp, unsigned short* __restrict__ xo) {
  const int N = 1024, H = 12, TC = 2304, C = 768;
  const float scale = 0.125f;
  __shared__ unsigned short ldsP[8][16 * 32];

  int gw   = (blockIdx.x * blockDim.x + threadIdx.x) >> 5;
  int lane = threadIdx.x & 31;
  int wv   = threadIdx.x >> 5;
  int qt   = gw & 63;
  int h    = (gw >> 6) % H;
  int b    = gw / (64 * H);
  int r = lane & 15, khalf = (lane >> 4) * 8;

  const unsigned short* qrow = qkv + (size_t)(b * N + qt * 16 + r) * TC + h * 64;
  v16bf qa0 = load_frag(qrow, khalf);
  v16bf qa1 = load_frag(qrow + 32, khalf);

  // V^T rows for this lane's n-position in each of the 4 output col tiles
  const unsigned short* vtb = vt + (size_t)(b * H + h) * 64 * N;
  const unsigned short* vr0 = vtb + (size_t)( 0 + r) * N;
  const unsigned short* vr1 = vtb + (size_t)(16 + r) * N;
  const unsigned short* vr2 = vtb + (size_t)(32 + r) * N;
  const unsigned short* vr3 = vtb + (size_t)(48 + r) * N;

  float lsev[8];
#pragma unroll
  for (int v = 0; v < 8; v++)
    lsev[v] = lse[((size_t)(b * H + h)) * N + qt * 16 + v + 8 * (lane >> 4)];

  v8f o0 = {}, o1 = {}, o2 = {}, o3 = {};
  for (int kb = 0; kb < 32; kb++) {
#pragma unroll
    for (int sub = 0; sub < 2; sub++) {
      int ks = kb * 32 + sub * 16;
      const unsigned short* krow = qkv + (size_t)(b * N + ks + r) * TC + 768 + h * 64;
      v8f c = {};
      c = wmma_bf16(qa0, load_frag(krow, khalf), c);
      c = wmma_bf16(qa1, load_frag(krow + 32, khalf), c);
      float lsz  = __logf(size[b * N + ks + r]);
      float tsum = 0.f;
#pragma unroll
      for (int v = 0; v < 8; v++) {
        float p = __expf(c[v] * scale + lsz - lsev[v]);
        tsum += p;
        ldsP[wv][(v + 8 * (lane >> 4)) * 32 + sub * 16 + r] = f2bf(p);
      }
      tsum += __shfl_xor(tsum, 16, 32);
      if (lane < 16) atomicAdd(&imp[b * N + ks + lane], tsum);
    }
    asm volatile("s_wait_dscnt 0" ::: "memory");   // LDS RAW fence (same wave)
    BFrag pf;
    const unsigned short* lrow = &ldsP[wv][r * 32];
    pf.q[0] = *(const uint4*)(lrow + khalf);        // ds_load_b128
    pf.q[1] = *(const uint4*)(lrow + khalf + 16);

    int kc = kb * 32;
    o0 = wmma_bf16(pf.v, load_frag(vr0 + kc, khalf), o0);
    o1 = wmma_bf16(pf.v, load_frag(vr1 + kc, khalf), o1);
    o2 = wmma_bf16(pf.v, load_frag(vr2 + kc, khalf), o2);
    o3 = wmma_bf16(pf.v, load_frag(vr3 + kc, khalf), o3);
  }
  v8f ov[4] = {o0, o1, o2, o3};
#pragma unroll
  for (int t = 0; t < 4; t++)
#pragma unroll
    for (int v = 0; v < 8; v++)
      xo[(size_t)(b * N + qt * 16 + v + 8 * (lane >> 4)) * C + h * 64 + t * 16 + r] =
          f2bf(ov[t][v]);
}

__global__ void k_impfin(float* imp) {             // /= H*N, set index0 = inf
  int i = blockIdx.x * blockDim.x + threadIdx.x;
  if (i < 8192) {
    float v = imp[i] * (1.0f / 12288.0f);
    if ((i & 1023) == 0) v = INFINITY;
    imp[i] = v;
  }
}

// metric = normalize(mean_h k)
__global__ void k_metric(const unsigned short* __restrict__ qkv, float* __restrict__ metric) {
  const int TC = 2304;
  int i = blockIdx.x * blockDim.x + threadIdx.x;   // b*1024 + n
  if (i >= 8192) return;
  const unsigned short* kp = qkv + (size_t)i * TC + 768;
  float m[64];
#pragma unroll
  for (int d = 0; d < 64; d++) {
    float s = 0.f;
    for (int h = 0; h < 12; h++) s += bf2f(kp[h * 64 + d]);
    m[d] = s * (1.0f / 12.0f);
  }
  float nrm = 0.f;
#pragma unroll
  for (int d = 0; d < 64; d++) nrm += m[d] * m[d];
  float inv = rsqrtf(nrm);
#pragma unroll
  for (int d = 0; d < 64; d++) metric[(size_t)i * 64 + d] = m[d] * inv;
}

// scores = a @ bm^T, row-max/argmax; row 0 forced to -inf
__global__ void k_nodes(const float* __restrict__ metric, float* __restrict__ node_max,
                        int* __restrict__ node_idx) {
  int i = blockIdx.x * blockDim.x + threadIdx.x;   // b*512 + s
  if (i >= 4096) return;
  int b = i >> 9, s = i & 511;
  if (s == 0) { node_max[i] = -INFINITY; node_idx[i] = 0; return; }
  float av[64];
  const float* a = metric + ((size_t)(b * 1024 + 2 * s)) * 64;
#pragma unroll
  for (int c = 0; c < 64; c++) av[c] = a[c];
  float best = -INFINITY; int bi = 0;
  for (int t = 0; t < 512; t++) {
    const float* bm = metric + ((size_t)(b * 1024 + 2 * t + 1)) * 64;
    float d = 0.f;
#pragma unroll
    for (int c = 0; c < 64; c++) d += av[c] * bm[c];
    if (d > best) { best = d; bi = t; }
  }
  node_max[i] = best; node_idx[i] = bi;
}

// top-r selection with jax top_k tie-break + stable-argsort partition
__global__ __launch_bounds__(512)
void k_merge_sel(const float* __restrict__ node_max, const int* __restrict__ node_idx,
                 int* __restrict__ unm_idx, int* __restrict__ merge_idx,
                 int* __restrict__ dst_idx) {
  int b = blockIdx.x, i = threadIdx.x;
  __shared__ float sv[512];
  __shared__ int   sc[512];
  sv[i] = node_max[b * 512 + i];
  __syncthreads();
  float vi = sv[i];
  int rank = 0;
  for (int j = 0; j < 512; j++) {
    float vj = sv[j];
    rank += (vj > vi) || (vj == vi && j < i);
  }
  int flag = rank < 128;
  sc[i] = flag;
  __syncthreads();
  for (int off = 1; off < 512; off <<= 1) {
    int add = (i >= off) ? sc[i - off] : 0;
    __syncthreads();
    sc[i] += add;
    __syncthreads();
  }
  int incl = sc[i];
  if (flag) {
    merge_idx[b * 128 + incl - 1] = i;
    dst_idx[b * 128 + incl - 1]   = node_idx[b * 512 + i];
  } else {
    unm_idx[b * 384 + (i - incl)] = i;
  }
}

// concat(unm_x, dst_x) with xw = x*size; importance carried along
__global__ void k_merge_init(const float* __restrict__ xres, const float* __restrict__ size_in,
                             const float* __restrict__ imp, const int* __restrict__ unm_idx,
                             float* __restrict__ xm, float* __restrict__ sm,
                             float* __restrict__ im) {
  int p = blockIdx.x % 896, b = blockIdx.x / 896, tid = threadIdx.x;
  int tok = (p < 384) ? 2 * unm_idx[b * 384 + p] : 2 * (p - 384) + 1;
  float sz = size_in[b * 1024 + tok];
  const float* xr = xres + ((size_t)(b * 1024 + tok)) * 768;
  float* xd = xm + ((size_t)(b * 896 + p)) * 768;
  for (int c = tid; c < 768; c += 128) xd[c] = xr[c] * sz;
  if (tid == 0) { sm[b * 896 + p] = sz; im[b * 896 + p] = imp[b * 1024 + tok]; }
}

__global__ void k_merge_scat(const float* __restrict__ xres, const float* __restrict__ size_in,
                             const float* __restrict__ imp, const int* __restrict__ merge_idx,
                             const int* __restrict__ dst_idx, float* __restrict__ xm,
                             float* __restrict__ sm, float* __restrict__ im) {
  int j = blockIdx.x & 127, b = blockIdx.x >> 7, tid = threadIdx.x;
  int s = merge_idx[b * 128 + j];
  int d = dst_idx[b * 128 + j];
  int tok = 2 * s;
  float sz = size_in[b * 1024 + tok];
  const float* xr = xres + ((size_t)(b * 1024 + tok)) * 768;
  float* xd = xm + ((size_t)(b * 896 + 384 + d)) * 768;
  for (int c = tid; c < 768; c += 128) atomicAdd(&xd[c], xr[c] * sz);
  if (tid == 0) {
    atomicAdd(&sm[b * 896 + 384 + d], sz);
    atomicMax((int*)&im[b * 896 + 384 + d], __float_as_int(imp[b * 1024 + tok])); // imp >= 0
  }
}

__global__ __launch_bounds__(896)
void k_keep_sel(const float* __restrict__ im, int* __restrict__ keep_idx) {
  int b = blockIdx.x, i = threadIdx.x;   // 896 threads
  __shared__ float sv[896];
  __shared__ int   sc[896];
  sv[i] = im[b * 896 + i];
  __syncthreads();
  float vi = sv[i];
  int rank = 0;
  for (int j = 0; j < 896; j++) {
    float vj = sv[j];
    rank += (vj > vi) || (vj == vi && j < i);
  }
  int flag = rank < 768;
  sc[i] = flag;
  __syncthreads();
  for (int off = 1; off < 896; off <<= 1) {
    int add = (i >= off) ? sc[i - off] : 0;
    __syncthreads();
    sc[i] += add;
    __syncthreads();
  }
  if (flag) keep_idx[b * 768 + sc[i] - flag] = i;  // exclusive prefix of kept
}

__global__ void k_keep_gather(const float* __restrict__ xm, const float* __restrict__ sm,
                              const int* __restrict__ keep_idx, float* __restrict__ xk,
                              float* __restrict__ size_out) {
  int p = blockIdx.x % 768, b = blockIdx.x / 768, tid = threadIdx.x;
  int src = keep_idx[b * 768 + p];
  float s = sm[b * 896 + src];
  float invs = 1.0f / s;
  const float* xr = xm + ((size_t)(b * 896 + src)) * 768;
  float* xd = xk + ((size_t)(b * 768 + p)) * 768;
  for (int c = tid; c < 768; c += 128) xd[c] = xr[c] * invs;
  if (tid == 0) size_out[b * 768 + p] = s;
}

// ---------------------------------------------------------------------------
// Host orchestration
// ---------------------------------------------------------------------------
extern "C" void kernel_launch(void* const* d_in, const int* in_sizes, int n_in,
                              void* d_out, int out_size, void* d_ws, size_t ws_size,
                              hipStream_t stream) {
  (void)in_sizes; (void)n_in; (void)out_size; (void)ws_size;
  const float* x      = (const float*)d_in[0];
  const float* size_  = (const float*)d_in[1];
  const float* w_qkv  = (const float*)d_in[2];
  const float* w_proj = (const float*)d_in[3];
  const float* b_proj = (const float*)d_in[4];
  const float* g1     = (const float*)d_in[5];
  const float* be1    = (const float*)d_in[6];
  const float* g2     = (const float*)d_in[7];
  const float* be2    = (const float*)d_in[8];
  const float* w_fc1  = (const float*)d_in[9];
  const float* b_fc1  = (const float*)d_in[10];
  const float* w_fc2  = (const float*)d_in[11];
  const float* b_fc2  = (const float*)d_in[12];

  char* ws = (char*)d_ws;
  size_t off = 0;
  auto alloc = [&](size_t bytes) -> void* {
    off = (off + 255) & ~(size_t)255;
    void* p = ws + off;
    off += bytes;
    return p;
  };
  unsigned short* wqkv_b  = (unsigned short*)alloc((size_t)2304 * 768 * 2);
  unsigned short* wproj_b = (unsigned short*)alloc((size_t)768 * 768 * 2);
  unsigned short* wfc1_b  = (unsigned short*)alloc((size_t)3072 * 768 * 2);
  unsigned short* wfc2_b  = (unsigned short*)alloc((size_t)768 * 3072 * 2);
  unsigned short* h1_b    = (unsigned short*)alloc((size_t)8192 * 768 * 2);   // reused as h2
  unsigned short* qkv_b   = (unsigned short*)alloc((size_t)8192 * 2304 * 2);  // reused as a1
  unsigned short* vt      = (unsigned short*)alloc((size_t)8 * 12 * 64 * 1024 * 2);
  float*          lse     = (float*)alloc((size_t)8 * 12 * 1024 * 4);
  unsigned short* xo_b    = (unsigned short*)alloc((size_t)8192 * 768 * 2);
  float*          imp     = (float*)alloc((size_t)8192 * 4);
  float*          xres    = (float*)alloc((size_t)8192 * 768 * 4);
  float*          metric  = (float*)alloc((size_t)8192 * 64 * 4);
  float*          nmax    = (float*)alloc((size_t)4096 * 4);
  int*            nidx    = (int*)alloc((size_t)4096 * 4);
  int*            unm_idx = (int*)alloc((size_t)8 * 384 * 4);
  int*            mrg_idx = (int*)alloc((size_t)8 * 128 * 4);
  int*            dst_idx = (int*)alloc((size_t)8 * 128 * 4);
  float*          xm      = (float*)alloc((size_t)8 * 896 * 768 * 4);
  float*          sm      = (float*)alloc((size_t)8 * 896 * 4);
  float*          im      = (float*)alloc((size_t)8 * 896 * 4);
  int*            keep_idx= (int*)alloc((size_t)8 * 768 * 4);
  float*          xk      = (float*)alloc((size_t)6144 * 768 * 4);

  float* outx    = (float*)d_out;
  float* outsize = outx + (size_t)8 * 768 * 768;

  // weights -> bf16
  k_f2b<<<(2304 * 768 + 255) / 256, 256, 0, stream>>>(w_qkv, wqkv_b, 2304 * 768);
  k_f2b<<<(768 * 768 + 255) / 256, 256, 0, stream>>>(w_proj, wproj_b, 768 * 768);
  k_f2b<<<(3072 * 768 + 255) / 256, 256, 0, stream>>>(w_fc1, wfc1_b, 3072 * 768);
  k_f2b<<<(768 * 3072 + 255) / 256, 256, 0, stream>>>(w_fc2, wfc2_b, 768 * 3072);

  // LN1 + QKV
  k_layernorm<<<8192, 256, 0, stream>>>(x, g1, be1, h1_b, 768);
  k_gemm<<<dim3(2304 / 64, 8192 / 128), 256, 0, stream>>>(
      h1_b, wqkv_b, 8192, 2304, 768, nullptr, nullptr, 0, nullptr, qkv_b);

  // attention
  k_vtrans<<<(8 * 12 * 64 * 1024) / 256, 256, 0, stream>>>(qkv_b, vt);
  k_attn_lse<<<768, 256, 0, stream>>>(qkv_b, size_, lse);
  k_zero<<<8192 / 256, 256, 0, stream>>>(imp, 8192);
  k_attn_av<<<768, 256, 0, stream>>>(qkv_b, vt, size_, lse, imp, xo_b);
  k_impfin<<<8192 / 256, 256, 0, stream>>>(imp);

  // proj + residual
  k_gemm<<<dim3(768 / 64, 8192 / 128), 256, 0, stream>>>(
      xo_b, wproj_b, 8192, 768, 768, b_proj, x, 0, xres, nullptr);

  // token merging
  k_metric<<<8192 / 256, 256, 0, stream>>>(qkv_b, metric);
  k_nodes<<<4096 / 256, 256, 0, stream>>>(metric, nmax, nidx);
  k_merge_sel<<<8, 512, 0, stream>>>(nmax, nidx, unm_idx, mrg_idx, dst_idx);
  k_merge_init<<<8 * 896, 128, 0, stream>>>(xres, size_, imp, unm_idx, xm, sm, im);
  k_merge_scat<<<8 * 128, 128, 0, stream>>>(xres, size_, imp, mrg_idx, dst_idx, xm, sm, im);

  // importance pruning
  k_keep_sel<<<8, 896, 0, stream>>>(im, keep_idx);
  k_keep_gather<<<8 * 768, 128, 0, stream>>>(xm, sm, keep_idx, xk, outsize);

  // MLP
  unsigned short* h2_b = h1_b;
  unsigned short* a1_b = qkv_b;
  k_layernorm<<<6144, 256, 0, stream>>>(xk, g2, be2, h2_b, 768);
  k_gemm<<<dim3(3072 / 64, 6144 / 128), 256, 0, stream>>>(
      h2_b, wfc1_b, 6144, 3072, 768, b_fc1, nullptr, 1, nullptr, a1_b);
  k_gemm<<<dim3(768 / 64, 6144 / 128), 256, 0, stream>>>(
      a1_b, wfc2_b, 6144, 768, 3072, b_fc2, xk, 0, outx, nullptr);
}
